// FRIEND_67680094650640
// MI455X (gfx1250) — matrix operand order repaired and checked by reading, not compiled, not run on hardware
//
#include <hip/hip_runtime.h>

#define HH 256
#define LL 50
#define SS 50
#define MAXF 32
#define BB 64
#define ROWS (BB * MAXF)   // 2048

typedef __attribute__((ext_vector_type(16))) __bf16 bf16x16;
typedef __attribute__((ext_vector_type(8)))  __bf16 bf16x8;
typedef __attribute__((ext_vector_type(8)))  float  f32x8;
typedef __attribute__((ext_vector_type(4)))  float  f32x4;

__device__ __forceinline__ bf16x16 cat_bf(bf16x8 lo, bf16x8 hi) {
  return __builtin_shufflevector(lo, hi, 0,1,2,3,4,5,6,7,8,9,10,11,12,13,14,15);
}
__device__ __forceinline__ bf16x8 cvt8(const float* p) {
  f32x4 a = *(const f32x4*)p;
  f32x4 b = *(const f32x4*)(p + 4);
  bf16x8 r;
  r[0]=(__bf16)a[0]; r[1]=(__bf16)a[1]; r[2]=(__bf16)a[2]; r[3]=(__bf16)a[3];
  r[4]=(__bf16)b[0]; r[5]=(__bf16)b[1]; r[6]=(__bf16)b[2]; r[7]=(__bf16)b[3];
  return r;
}
__device__ __forceinline__ f32x8 wmma_bf(bf16x16 a, bf16x16 b, f32x8 c) {
  return __builtin_amdgcn_wmma_f32_16x16x32_bf16(false, a, false, b, (short)0, c, false, false);
}

// ---------------------------------------------------------------------------
// Prep: convert weights fp32->bf16 (Wb transposed), zero initial states.
// ---------------------------------------------------------------------------
__global__ __launch_bounds__(256)
void k_prep(const float* __restrict__ Wih0, const float* __restrict__ Whh0,
            const float* __restrict__ Wih1, const float* __restrict__ Whh1,
            const float* __restrict__ Wf,   const float* __restrict__ Wb,
            __bf16* wih0b, __bf16* whh0b, __bf16* wih1b, __bf16* whh1b,
            __bf16* wfb, __bf16* wbtb,
            float* h0f, __bf16* h0b, float* h1f, __bf16* h1b,
            float* flf, __bf16* flb)
{
  unsigned i = blockIdx.x * 256 + threadIdx.x;
  if (i < 3 * HH * HH) {
    wih0b[i] = (__bf16)Wih0[i]; whh0b[i] = (__bf16)Whh0[i];
    wih1b[i] = (__bf16)Wih1[i]; whh1b[i] = (__bf16)Whh1[i];
  }
  if (i < HH * 2 * HH) wfb[i] = (__bf16)Wf[i];
  if (i < HH * HH) {                      // WbT[h][g] = Wb[g][h]
    unsigned g = i >> 8, h = i & 255;
    wbtb[h * HH + g] = (__bf16)Wb[i];
  }
  if (i < ROWS * HH) {
    h0f[i] = 0.f; h1f[i] = 0.f; flf[i] = 0.f;
    h0b[i] = (__bf16)0.f; h1b[i] = (__bf16)0.f; flb[i] = (__bf16)0.f;
  }
}

// ---------------------------------------------------------------------------
// One fused GRU step (gi GEMM + gh GEMM + gate math) for one layer.
// Grid: (8 col-tiles of 32, 32 row-tiles of 64), 256 threads = 8 waves.
// Each wave: 16x16 tile for each of 6 gate accumulators (3 gates x {gi,gh}).
// XB16 = true: x is contiguous bf16 [ROWS][HH]; false: x is strided fp32.
// ---------------------------------------------------------------------------
template<bool XB16>
__global__ __launch_bounds__(256)
void k_gru_step(const float* __restrict__ xf, unsigned xstride,
                const __bf16* __restrict__ xbf,
                const __bf16* __restrict__ hprev_bf,
                const float*  __restrict__ hprev_f,
                const __bf16* __restrict__ Wih, const __bf16* __restrict__ Whh,
                const float* __restrict__ bih, const float* __restrict__ bhh,
                float* __restrict__ hnext_f, __bf16* __restrict__ hnext_bf,
                int t, const int* __restrict__ flen,
                float* __restrict__ flf, __bf16* __restrict__ flb)
{
  const unsigned lane = threadIdx.x & 31;
  const unsigned wave = threadIdx.x >> 5;
  const unsigned r0 = blockIdx.y * 64 + (wave & 3) * 16;
  const unsigned c0 = blockIdx.x * 32 + (wave >> 2) * 16;
  const unsigned am  = r0 + (lane & 15);
  const unsigned akb = (lane >> 4) * 16;
  const unsigned bn  = lane & 15;
  const unsigned bko = (lane >> 4) * 16;

  const unsigned aoff_h = am * HH + akb;            // bf16 elements
  const unsigned aoff_x = XB16 ? aoff_h : (am * xstride + akb);
  const unsigned woff0 = (0 * HH + c0 + bn) * HH + bko;
  const unsigned woff1 = (1 * HH + c0 + bn) * HH + bko;
  const unsigned woff2 = (2 * HH + c0 + bn) * HH + bko;

  f32x8 gi[3] = {f32x8{}, f32x8{}, f32x8{}};
  f32x8 gh[3] = {f32x8{}, f32x8{}, f32x8{}};

  for (unsigned k0 = 0; k0 < HH; k0 += 32) {
    bf16x16 ax, ah;
    if (XB16) {
      const __bf16* p = xbf + (aoff_x + k0);
      ax = cat_bf(*(const bf16x8*)p, *(const bf16x8*)(p + 8));
    } else {
      const float* p = xf + (aoff_x + k0);
      ax = cat_bf(cvt8(p), cvt8(p + 8));
    }
    {
      const __bf16* p = hprev_bf + (aoff_h + k0);
      ah = cat_bf(*(const bf16x8*)p, *(const bf16x8*)(p + 8));
    }
    {
      bf16x16 bi = *(const bf16x16*)(Wih + (woff0 + k0));
      bf16x16 bh = *(const bf16x16*)(Whh + (woff0 + k0));
      gi[0] = wmma_bf(ax, bi, gi[0]);
      gh[0] = wmma_bf(ah, bh, gh[0]);
    }
    {
      bf16x16 bi = *(const bf16x16*)(Wih + (woff1 + k0));
      bf16x16 bh = *(const bf16x16*)(Whh + (woff1 + k0));
      gi[1] = wmma_bf(ax, bi, gi[1]);
      gh[1] = wmma_bf(ah, bh, gh[1]);
    }
    {
      bf16x16 bi = *(const bf16x16*)(Wih + (woff2 + k0));
      bf16x16 bh = *(const bf16x16*)(Whh + (woff2 + k0));
      gi[2] = wmma_bf(ax, bi, gi[2]);
      gh[2] = wmma_bf(ah, bh, gh[2]);
    }
  }

  const unsigned n = c0 + bn;
  const float bir = bih[n], biz = bih[HH + n], bin_ = bih[2 * HH + n];
  const float bhr = bhh[n], bhz = bhh[HH + n], bhn  = bhh[2 * HH + n];
  const unsigned mt = r0 + ((lane >> 4) << 3);
#pragma unroll
  for (int v = 0; v < 8; ++v) {
    const unsigned m = mt + v;
    const unsigned o = m * HH + n;
    const float rr = 1.f / (1.f + __expf(-(gi[0][v] + bir + gh[0][v] + bhr)));
    const float zz = 1.f / (1.f + __expf(-(gi[1][v] + biz + gh[1][v] + bhz)));
    const float nn = tanhf(gi[2][v] + bin_ + rr * (gh[2][v] + bhn));
    const float hp = hprev_f[o];
    const float hN = (1.f - zz) * nn + zz * hp;
    hnext_f [o] = hN;
    hnext_bf[o] = (__bf16)hN;
    if (flf) {
      if (t == flen[m] - 1) {
        flf[o] = hN;
        flb[o] = (__bf16)hN;
      }
    }
  }
}

// ---------------------------------------------------------------------------
// sf = concat(self_x[b], friend_last[row]) @ Wf^T   (M=2048,K=512,N=256) -> bf16
// ---------------------------------------------------------------------------
__global__ __launch_bounds__(256)
void k_sf(const float* __restrict__ self_x, const __bf16* __restrict__ flb,
          const __bf16* __restrict__ wfb, __bf16* __restrict__ sfb)
{
  const unsigned lane = threadIdx.x & 31;
  const unsigned wave = threadIdx.x >> 5;
  const unsigned r0 = blockIdx.y * 64 + (wave & 3) * 16;
  const unsigned c0 = blockIdx.x * 32 + (wave >> 2) * 16;
  const unsigned am  = r0 + (lane & 15);
  const unsigned b   = am >> 5;
  const unsigned akb = (lane >> 4) * 16;
  const unsigned bn  = lane & 15;
  const unsigned bko = (lane >> 4) * 16;

  const unsigned soff = b * HH + akb;
  const unsigned foff = am * HH + akb;
  const unsigned woff = (c0 + bn) * (2 * HH) + bko;

  f32x8 acc = {};
  // K-halves never straddle: k0 multiple of 32, boundary at 256.
#pragma unroll 1
  for (unsigned k0 = 0; k0 < HH; k0 += 32) {      // self_x half (fp32 -> bf16)
    bf16x16 a = cat_bf(cvt8(self_x + soff + k0), cvt8(self_x + soff + k0 + 8));
    bf16x16 bb = *(const bf16x16*)(wfb + woff + k0);
    acc = wmma_bf(a, bb, acc);
  }
#pragma unroll 1
  for (unsigned k0 = 0; k0 < HH; k0 += 32) {      // friend_last half (bf16)
    const __bf16* p = flb + foff + k0;
    bf16x16 a = cat_bf(*(const bf16x8*)p, *(const bf16x8*)(p + 8));
    bf16x16 bb = *(const bf16x16*)(wfb + woff + HH + k0);
    acc = wmma_bf(a, bb, acc);
  }
  const unsigned n = c0 + bn;
  const unsigned mt = r0 + ((lane >> 4) << 3);
#pragma unroll
  for (int v = 0; v < 8; ++v)
    sfb[(mt + v) * HH + n] = (__bf16)acc[v];
}

// ---------------------------------------------------------------------------
// v = sf @ Wb   (uses WbT row-major so B-frag loads are contiguous) -> fp32
// ---------------------------------------------------------------------------
__global__ __launch_bounds__(256)
void k_v(const __bf16* __restrict__ sfb, const __bf16* __restrict__ wbtb,
         float* __restrict__ vout)
{
  const unsigned lane = threadIdx.x & 31;
  const unsigned wave = threadIdx.x >> 5;
  const unsigned r0 = blockIdx.y * 64 + (wave & 3) * 16;
  const unsigned c0 = blockIdx.x * 32 + (wave >> 2) * 16;
  const unsigned am  = r0 + (lane & 15);
  const unsigned akb = (lane >> 4) * 16;
  const unsigned bn  = lane & 15;
  const unsigned bko = (lane >> 4) * 16;

  const unsigned aoff = am * HH + akb;
  const unsigned woff = (c0 + bn) * HH + bko;

  f32x8 acc = {};
  for (unsigned k0 = 0; k0 < HH; k0 += 32) {
    const __bf16* p = sfb + aoff + k0;
    bf16x16 a = cat_bf(*(const bf16x8*)p, *(const bf16x8*)(p + 8));
    bf16x16 b = *(const bf16x16*)(wbtb + woff + k0);
    acc = wmma_bf(a, b, acc);
  }
  const unsigned n = c0 + bn;
  const unsigned mt = r0 + ((lane >> 4) << 3);
#pragma unroll
  for (int v = 0; v < 8; ++v)
    vout[(mt + v) * HH + n] = acc[v];
}

// ---------------------------------------------------------------------------
// Attention: tf = sum_s softplus(common_x . v) * exp(-time) * mask;
// softmax over padded MAXF; out = sum_f w * friend_last.  One block per b.
// ---------------------------------------------------------------------------
__global__ __launch_bounds__(256)
void k_attn(const float* __restrict__ common_x, const float* __restrict__ common_time,
            const int* __restrict__ clen, const int* __restrict__ fnum,
            const float* __restrict__ vv, const float* __restrict__ flf,
            float* __restrict__ out)
{
  const unsigned b = blockIdx.x;
  const unsigned lane = threadIdx.x & 31;
  const unsigned wave = threadIdx.x >> 5;
  __shared__ float tf_s[MAXF];
  const int fn = fnum[b];

  for (unsigned f = wave; f < MAXF; f += 8) {
    const unsigned row = b * MAXF + f;
    const unsigned voff = row * HH;
    float vl[8];
#pragma unroll
    for (int j = 0; j < 8; ++j) vl[j] = vv[voff + j * 32 + lane];
    const unsigned cxoff = row * (SS * HH);
    const unsigned ctoff = row * SS;
    const int cl = clen[row];
    float acc = 0.f;
    for (int s = 0; s < SS; ++s) {
      const unsigned co = cxoff + (unsigned)s * HH;
      float p = 0.f;
#pragma unroll
      for (int j = 0; j < 8; ++j) p += common_x[co + j * 32 + lane] * vl[j];
#pragma unroll
      for (int off = 16; off; off >>= 1) p += __shfl_xor(p, off, 32);
      const float sp = (p > 20.f) ? p : log1pf(__expf(p));
      const float wgt = (s < cl) ? __expf(-common_time[ctoff + s]) : 0.f;
      acc += sp * wgt;
    }
    if (lane == 0) tf_s[f] = ((int)f < fn) ? acc : 0.f;
  }
  __syncthreads();

  const unsigned h = threadIdx.x;
  float mx = -3.4e38f;
  for (int f = 0; f < MAXF; ++f) mx = fmaxf(mx, tf_s[f]);
  float Z = 0.f;
  for (int f = 0; f < MAXF; ++f) Z += __expf(tf_s[f] - mx);
  const float inv = 1.f / Z;
  float o = 0.f;
  for (int f = 0; f < MAXF; ++f) {
    const float w = __expf(tf_s[f] - mx) * inv;
    const float flv = (f < fn) ? flf[(b * MAXF + f) * HH + h] : 0.f;
    o += w * flv;
  }
  out[b * HH + h] = o;
}

// ---------------------------------------------------------------------------
extern "C" void kernel_launch(void* const* d_in, const int* in_sizes, int n_in,
                              void* d_out, int out_size, void* d_ws, size_t ws_size,
                              hipStream_t stream)
{
  const float* self_x   = (const float*)d_in[0];
  const float* common_x = (const float*)d_in[1];
  const float* common_t = (const float*)d_in[2];
  const float* friend_x = (const float*)d_in[3];
  const float* Wih0 = (const float*)d_in[4];
  const float* Whh0 = (const float*)d_in[5];
  const float* bih0 = (const float*)d_in[6];
  const float* bhh0 = (const float*)d_in[7];
  const float* Wih1 = (const float*)d_in[8];
  const float* Whh1 = (const float*)d_in[9];
  const float* bih1 = (const float*)d_in[10];
  const float* bhh1 = (const float*)d_in[11];
  const float* Wf   = (const float*)d_in[12];
  const float* Wb   = (const float*)d_in[13];
  const int* flen = (const int*)d_in[14];
  const int* fnum = (const int*)d_in[15];
  const int* clen = (const int*)d_in[16];
  float* out = (float*)d_out;

  char* ws = (char*)d_ws;
  size_t off = 0;
  auto alloc = [&](size_t bytes) -> char* {
    char* p = ws + off;
    off += (bytes + 255) & ~(size_t)255;
    return p;
  };

  __bf16* wih0b = (__bf16*)alloc(3 * HH * HH * 2);
  __bf16* whh0b = (__bf16*)alloc(3 * HH * HH * 2);
  __bf16* wih1b = (__bf16*)alloc(3 * HH * HH * 2);
  __bf16* whh1b = (__bf16*)alloc(3 * HH * HH * 2);
  __bf16* wfb   = (__bf16*)alloc(HH * 2 * HH * 2);
  __bf16* wbtb  = (__bf16*)alloc(HH * HH * 2);
  float*  h0f[2]; __bf16* h0b[2]; float* h1f[2]; __bf16* h1b[2];
  for (int i = 0; i < 2; ++i) {
    h0f[i] = (float*) alloc((size_t)ROWS * HH * 4);
    h0b[i] = (__bf16*)alloc((size_t)ROWS * HH * 2);
    h1f[i] = (float*) alloc((size_t)ROWS * HH * 4);
    h1b[i] = (__bf16*)alloc((size_t)ROWS * HH * 2);
  }
  float*  flf = (float*) alloc((size_t)ROWS * HH * 4);
  __bf16* flb = (__bf16*)alloc((size_t)ROWS * HH * 2);
  __bf16* sfb = (__bf16*)alloc((size_t)ROWS * HH * 2);
  float*  vf  = (float*) alloc((size_t)ROWS * HH * 4);

  k_prep<<<(ROWS * HH) / 256, 256, 0, stream>>>(
      Wih0, Whh0, Wih1, Whh1, Wf, Wb,
      wih0b, whh0b, wih1b, whh1b, wfb, wbtb,
      h0f[0], h0b[0], h1f[0], h1b[0], flf, flb);

  const dim3 gemm_grid(8, 32, 1);
  for (int t = 0; t < LL; ++t) {
    const int pi = t & 1, ni = pi ^ 1;
    // Layer 0: input = friend_x[:, :, t, :] (row stride L*H), strided fp32
    k_gru_step<false><<<gemm_grid, 256, 0, stream>>>(
        friend_x + (size_t)t * HH, (unsigned)(LL * HH), nullptr,
        h0b[pi], h0f[pi], wih0b, whh0b, bih0, bhh0,
        h0f[ni], h0b[ni], t, nullptr, nullptr, nullptr);
    // Layer 1: input = layer-0 output just produced (bf16, contiguous)
    k_gru_step<true><<<gemm_grid, 256, 0, stream>>>(
        nullptr, 0u, h0b[ni],
        h1b[pi], h1f[pi], wih1b, whh1b, bih1, bhh1,
        h1f[ni], h1b[ni], t, flen, flf, flb);
  }

  k_sf<<<gemm_grid, 256, 0, stream>>>(self_x, flb, wfb, sfb);
  k_v <<<gemm_grid, 256, 0, stream>>>(sfb, wbtb, vf);
  k_attn<<<BB, 256, 0, stream>>>(common_x, common_t, clen, fnum, vf, flf, out);

  (void)in_sizes; (void)n_in; (void)out_size; (void)ws_size;
}
